// GNN_1_38001870635841
// MI455X (gfx1250) — matrix operand order repaired and checked by reading, not compiled, not run on hardware
//
#include <hip/hip_runtime.h>

// ---------------------------------------------------------------------------
// GNN forward for MI455X (gfx1250, wave32, WMMA f16 -> f32 accumulate)
// Register-blocked 1x4 (GEMM2) / 1x2 (GEMM3) WMMA tiles, conflict-free LDS.
// ---------------------------------------------------------------------------

typedef __attribute__((ext_vector_type(16))) _Float16 v16h;
typedef __attribute__((ext_vector_type(8)))  float    v8f;

#define ZD    64
#define AD    8
#define KN    64
#define HD    512
#define BATCH 64
#define NROWS (BATCH * KN)   // 4096 node rows
#define F2    160            // node-input features padded 136 -> 160 (mult of 32)
#define LDH   520            // LDS row stride in halfs: 260 words -> 4-bank shift
                             // per row -> conflict-free A-fragment reads

union F16x16 { v16h v; float4 q[2]; };

__device__ __forceinline__ v8f wmma16(v16h a, v16h b, v8f c) {
  return __builtin_amdgcn_wmma_f32_16x16x32_f16(false, a, false, b, (short)0, c,
                                                false, false);
}

// A-fragment (16x32 f16, row-major, ld in halfs). ISA 7.12.2 layout.
__device__ __forceinline__ v16h load_frag_a(const _Float16* base, int ld,
                                            int row0, int k0, int lane) {
  const int row = row0 + (lane & 15);
  const int col = k0 + ((lane >> 4) << 3);
  const _Float16* p = base + (size_t)row * ld + col;
  F16x16 f;
  f.q[0] = *reinterpret_cast<const float4*>(p);
  f.q[1] = *reinterpret_cast<const float4*>(p + 16);
  return f.v;
}

// B-fragment (32x16 f16) from transposed weights W[N][K] row-major (ld = K).
__device__ __forceinline__ v16h load_frag_b(const _Float16* base, int ld,
                                            int n0, int k0, int lane) {
  const int row = n0 + (lane & 15);
  const int col = k0 + ((lane >> 4) << 3);
  const _Float16* p = base + (size_t)row * ld + col;
  F16x16 f;
  f.q[0] = *reinterpret_cast<const float4*>(p);
  f.q[1] = *reinterpret_cast<const float4*>(p + 16);
  return f.v;
}

// Blocked GEMM slab: M=64, N=512, K=KT*32. 8 waves; wave (w) owns M-tile w/2
// and 4 groups of 4 N-tiles (parity w&1). relu(A@Bt^T + bias) -> dst (f16).
template <int KT>
__device__ __forceinline__ void gemm_block_relu(const _Float16* __restrict__ A,
                                                int lda,
                                                const _Float16* __restrict__ Bt,
                                                int ldb,
                                                const float* __restrict__ bias,
                                                _Float16* __restrict__ dst,
                                                int ldd, int w, int lane) {
  const int mt = w >> 1;
  for (int tg = (w & 1); tg < 8; tg += 2) {
    v8f acc[4] = {{}, {}, {}, {}};
#pragma unroll 2
    for (int kt = 0; kt < KT; ++kt) {
      v16h av = load_frag_a(A, lda, mt * 16, kt * 32, lane);
#pragma unroll
      for (int q = 0; q < 4; ++q) {
        v16h bf = load_frag_b(Bt, ldb, (tg * 4 + q) * 16, kt * 32, lane);
        acc[q] = wmma16(av, bf, acc[q]);
      }
    }
    const int rbase = mt * 16 + ((lane >> 4) << 3);
#pragma unroll
    for (int q = 0; q < 4; ++q) {
      const int col = (tg * 4 + q) * 16 + (lane & 15);
      const float bb = bias[col];
#pragma unroll
      for (int r = 0; r < 8; ++r) {
        float v = acc[q][r] + bb;
        dst[(rbase + r) * ldd + col] = (_Float16)(v > 0.0f ? v : 0.0f);
      }
    }
  }
}

// ---------------------------------------------------------------------------
// Prep kernels
// ---------------------------------------------------------------------------

__global__ void transpose_w(const float* __restrict__ in, int Cin, int r0,
                            int Rsub, int ldo, _Float16* __restrict__ out,
                            int total) {
  int idx = blockIdx.x * blockDim.x + threadIdx.x;
  if (idx >= total) return;
  int c = idx / ldo;
  int r = idx - c * ldo;
  out[idx] = (r < Rsub) ? (_Float16)in[(size_t)(r0 + r) * Cin + c]
                        : (_Float16)0.0f;
}

__global__ void conv_f16(const float* __restrict__ in, _Float16* __restrict__ out,
                         int n) {
  int idx = blockIdx.x * blockDim.x + threadIdx.x;
  if (idx < n) out[idx] = (_Float16)in[idx];
}

__global__ void build_x2(const float* __restrict__ x, const float* __restrict__ a,
                         const float* __restrict__ e, _Float16* __restrict__ X2) {
  int idx = blockIdx.x * blockDim.x + threadIdx.x;
  if (idx >= NROWS * F2) return;
  int row = idx / F2;
  int c   = idx - row * F2;
  float v;
  if      (c < 64)  v = x[(size_t)row * 64 + c];
  else if (c < 72)  v = a[(size_t)row * 8 + (c - 64)];
  else if (c < 136) v = e[(size_t)row * 64 + (c - 72)];
  else              v = 0.0f;
  X2[idx] = (_Float16)v;
}

// ---------------------------------------------------------------------------
// P = x @ W1a + be1 ; Q = x @ W1b     (M=4096, K=64, N=512), WMMA
// ---------------------------------------------------------------------------
__global__ __launch_bounds__(256) void pq_kernel(
    const _Float16* __restrict__ xh, const _Float16* __restrict__ W1a_t,
    const _Float16* __restrict__ W1b_t, const float* __restrict__ be1,
    _Float16* __restrict__ P, _Float16* __restrict__ Qm) {
  const int lane = threadIdx.x & 31;
  const int gw   = blockIdx.x * 8 + (threadIdx.x >> 5);  // 0..8191
  const int mt   = gw >> 5;   // 0..255
  const int nt   = gw & 31;   // 0..31

  v16h a0 = load_frag_a(xh, ZD, mt * 16, 0,  lane);
  v16h a1 = load_frag_a(xh, ZD, mt * 16, 32, lane);

  v8f accP = {}, accQ = {};
  accP = wmma16(a0, load_frag_b(W1a_t, ZD, nt * 16, 0,  lane), accP);
  accP = wmma16(a1, load_frag_b(W1a_t, ZD, nt * 16, 32, lane), accP);
  accQ = wmma16(a0, load_frag_b(W1b_t, ZD, nt * 16, 0,  lane), accQ);
  accQ = wmma16(a1, load_frag_b(W1b_t, ZD, nt * 16, 32, lane), accQ);

  const int col   = nt * 16 + (lane & 15);
  const int rbase = mt * 16 + ((lane >> 4) << 3);
  const float bias = be1[col];
#pragma unroll
  for (int r = 0; r < 8; ++r) {
    P [(size_t)(rbase + r) * HD + col] = (_Float16)(accP[r] + bias);
    Qm[(size_t)(rbase + r) * HD + col] = (_Float16)(accQ[r]);
  }
}

// ---------------------------------------------------------------------------
// Edge MLP: one block per (b,i). h1=relu(P_i+Q_j) -> h2=relu(h1@W2+b2)
// -> out=h2@W3, deterministic masked row-sum over j!=i, e = sum + 63*be3
// ---------------------------------------------------------------------------
__global__ __launch_bounds__(256) void edge_kernel(
    const _Float16* __restrict__ P, const _Float16* __restrict__ Q,
    const _Float16* __restrict__ W2t, const _Float16* __restrict__ W3t,
    const float* __restrict__ be2, const float* __restrict__ be3,
    float* __restrict__ e_out) {
  __shared__ __align__(16) _Float16 h1[64 * LDH];
  __shared__ __align__(16) _Float16 h2[64 * LDH];
  __shared__ __align__(16) _Float16 Pi[HD];
  __shared__ float part[8][64];

  const int wg   = blockIdx.x;   // b*64 + i
  const int i    = wg & 63;
  const int b    = wg >> 6;
  const int tid  = threadIdx.x;
  const int lane = tid & 31;
  const int w    = tid >> 5;     // wave 0..7

  const _Float16* prow = P + (size_t)wg * HD;
  for (int c = tid; c < HD; c += 256) Pi[c] = prow[c];
  __syncthreads();

  // h1[j][:] = relu(Pi + Q[b*64+j]), j = 0..63 (self row computed, masked later)
  const _Float16* qbase = Q + (size_t)b * KN * HD;
  for (int idx = tid; idx < KN * HD / 8; idx += 256) {
    int j  = idx >> 6;
    int cc = (idx & 63) << 3;
    union { float4 f; _Float16 h[8]; } uq, up, ur;
    uq.f = *reinterpret_cast<const float4*>(qbase + (size_t)j * HD + cc);
    up.f = *reinterpret_cast<const float4*>(Pi + cc);
#pragma unroll
    for (int t = 0; t < 8; ++t) {
      float v = (float)uq.h[t] + (float)up.h[t];
      ur.h[t] = (_Float16)(v > 0.0f ? v : 0.0f);
    }
    *reinterpret_cast<float4*>(&h1[j * LDH + cc]) = ur.f;
  }
  __syncthreads();

  // GEMM2: h2 = relu(h1 @ W2 + be2), M=64 N=512 K=512, 1x4 blocked
  gemm_block_relu<16>(h1, LDH, W2t, HD, be2, h2, LDH, w, lane);
  __syncthreads();

  // GEMM3: out = h2 @ W3 (M=64 N=64 K=512), 1x2 blocked; sum rows j != i
  {
    const int mt2 = w >> 1;
    const int ng  = (w & 1) * 2;
    v8f acc[2] = {{}, {}};
#pragma unroll 2
    for (int kt = 0; kt < 16; ++kt) {
      v16h av = load_frag_a(h2, LDH, mt2 * 16, kt * 32, lane);
      acc[0] = wmma16(av, load_frag_b(W3t, HD, (ng + 0) * 16, kt * 32, lane), acc[0]);
      acc[1] = wmma16(av, load_frag_b(W3t, HD, (ng + 1) * 16, kt * 32, lane), acc[1]);
    }
    const int rbase = mt2 * 16 + ((lane >> 4) << 3);
#pragma unroll
    for (int q = 0; q < 2; ++q) {
      const int col = (ng + q) * 16 + (lane & 15);
      float p = 0.0f;
#pragma unroll
      for (int r = 0; r < 8; ++r)
        if (rbase + r != i) p += acc[q][r];
      part[mt2 * 2 + (lane >> 4)][col] = p;  // unique slot per (Mtile, half, col)
    }
  }
  __syncthreads();

  if (tid < 64) {
    float s = 63.0f * be3[tid];  // bias once per summed edge
#pragma unroll
    for (int u = 0; u < 8; ++u) s += part[u][tid];
    e_out[(size_t)wg * 64 + tid] = s;
  }
}

// ---------------------------------------------------------------------------
// Node MLP: one block per 64-row slab (64 blocks). X2(160) -> 512 -> 512 -> 64
// ---------------------------------------------------------------------------
__global__ __launch_bounds__(256) void node_kernel(
    const _Float16* __restrict__ X2, const _Float16* __restrict__ Wn1t,
    const _Float16* __restrict__ Wn2t, const _Float16* __restrict__ Wn3t,
    const float* __restrict__ bn1, const float* __restrict__ bn2,
    const float* __restrict__ bn3, float* __restrict__ out) {
  __shared__ __align__(16) _Float16 h1[64 * LDH];
  __shared__ __align__(16) _Float16 h2[64 * LDH];

  const int rb   = blockIdx.x;  // rows rb*64 .. rb*64+63
  const int tid  = threadIdx.x;
  const int lane = tid & 31;
  const int w    = tid >> 5;

  const _Float16* Abase = X2 + (size_t)rb * 64 * F2;

  // GEMM1: h1 = relu(X2 @ Wn1 + bn1), M=64 N=512 K=160, 1x4 blocked
  gemm_block_relu<5>(Abase, F2, Wn1t, F2, bn1, h1, LDH, w, lane);
  __syncthreads();

  // GEMM2: h2 = relu(h1 @ Wn2 + bn2), M=64 N=512 K=512, 1x4 blocked
  gemm_block_relu<16>(h1, LDH, Wn2t, HD, bn2, h2, LDH, w, lane);
  __syncthreads();

  // GEMM3: out = h2 @ Wn3 + bn3, M=64 N=64 K=512, 1x2 blocked, f32 store
  {
    const int mt2 = w >> 1;
    const int ng  = (w & 1) * 2;
    v8f acc[2] = {{}, {}};
#pragma unroll 2
    for (int kt = 0; kt < 16; ++kt) {
      v16h av = load_frag_a(h2, LDH, mt2 * 16, kt * 32, lane);
      acc[0] = wmma16(av, load_frag_b(Wn3t, HD, (ng + 0) * 16, kt * 32, lane), acc[0]);
      acc[1] = wmma16(av, load_frag_b(Wn3t, HD, (ng + 1) * 16, kt * 32, lane), acc[1]);
    }
    const int rbase = mt2 * 16 + ((lane >> 4) << 3);
#pragma unroll
    for (int q = 0; q < 2; ++q) {
      const int col = (ng + q) * 16 + (lane & 15);
      const float bias = bn3[col];
#pragma unroll
      for (int r = 0; r < 8; ++r)
        out[(size_t)(rb * 64 + rbase + r) * 64 + col] = acc[q][r] + bias;
    }
  }
}

// ---------------------------------------------------------------------------
extern "C" void kernel_launch(void* const* d_in, const int* in_sizes, int n_in,
                              void* d_out, int out_size, void* d_ws, size_t ws_size,
                              hipStream_t stream) {
  const float* x   = (const float*)d_in[0];
  const float* a   = (const float*)d_in[1];
  const float* We1 = (const float*)d_in[2];
  const float* be1 = (const float*)d_in[3];
  const float* We2 = (const float*)d_in[4];
  const float* be2 = (const float*)d_in[5];
  const float* We3 = (const float*)d_in[6];
  const float* be3 = (const float*)d_in[7];
  const float* Wn1 = (const float*)d_in[8];
  const float* bn1 = (const float*)d_in[9];
  const float* Wn2 = (const float*)d_in[10];
  const float* bn2 = (const float*)d_in[11];
  const float* Wn3 = (const float*)d_in[12];
  const float* bn3 = (const float*)d_in[13];
  float* outp = (float*)d_out;

  char* ws = (char*)d_ws;
  size_t off = 0;
  auto alloc = [&](size_t bytes) -> void* {
    void* p = ws + off;
    off += (bytes + 255) & ~(size_t)255;
    return p;
  };

  _Float16* W1a_t = (_Float16*)alloc((size_t)HD * ZD * 2);   // [512][64]
  _Float16* W1b_t = (_Float16*)alloc((size_t)HD * ZD * 2);   // [512][64]
  _Float16* W2t   = (_Float16*)alloc((size_t)HD * HD * 2);   // [512][512]
  _Float16* W3t   = (_Float16*)alloc((size_t)ZD * HD * 2);   // [64][512]
  _Float16* Wn1t  = (_Float16*)alloc((size_t)HD * F2 * 2);   // [512][160] padded
  _Float16* Wn2t  = (_Float16*)alloc((size_t)HD * HD * 2);   // [512][512]
  _Float16* Wn3t  = (_Float16*)alloc((size_t)ZD * HD * 2);   // [64][512]
  _Float16* xh    = (_Float16*)alloc((size_t)NROWS * ZD * 2);
  _Float16* Pm    = (_Float16*)alloc((size_t)NROWS * HD * 2);
  _Float16* Qm    = (_Float16*)alloc((size_t)NROWS * HD * 2);
  float*    e_s   = (float*)   alloc((size_t)NROWS * ZD * 4);
  _Float16* X2    = (_Float16*)alloc((size_t)NROWS * F2 * 2);
  (void)ws_size; (void)in_sizes; (void)n_in; (void)out_size;

  const int TB = 256;
  auto blocks = [](int n) { return (n + 255) / 256; };

  int n;
  n = HD * ZD; transpose_w<<<blocks(n), TB, 0, stream>>>(We1, HD, 0,   ZD,  ZD, W1a_t, n);
  n = HD * ZD; transpose_w<<<blocks(n), TB, 0, stream>>>(We1, HD, 64,  ZD,  ZD, W1b_t, n);
  n = HD * HD; transpose_w<<<blocks(n), TB, 0, stream>>>(We2, HD, 0,   HD,  HD, W2t,  n);
  n = ZD * HD; transpose_w<<<blocks(n), TB, 0, stream>>>(We3, ZD, 0,   HD,  HD, W3t,  n);
  n = HD * F2; transpose_w<<<blocks(n), TB, 0, stream>>>(Wn1, HD, 0,   136, F2, Wn1t, n);
  n = HD * HD; transpose_w<<<blocks(n), TB, 0, stream>>>(Wn2, HD, 0,   HD,  HD, Wn2t, n);
  n = ZD * HD; transpose_w<<<blocks(n), TB, 0, stream>>>(Wn3, ZD, 0,   HD,  HD, Wn3t, n);

  n = NROWS * ZD; conv_f16<<<blocks(n), TB, 0, stream>>>(x, xh, n);

  pq_kernel<<<(NROWS / 16) * (HD / 16) / 8, TB, 0, stream>>>(xh, W1a_t, W1b_t,
                                                             be1, Pm, Qm);

  edge_kernel<<<NROWS, TB, 0, stream>>>(Pm, Qm, W2t, W3t, be2, be3, e_s);

  n = NROWS * F2; build_x2<<<blocks(n), TB, 0, stream>>>(x, a, e_s, X2);

  node_kernel<<<NROWS / 64, TB, 0, stream>>>(X2, Wn1t, Wn2t, Wn3t, bn1, bn2,
                                             bn3, outp);
}
